// relRnn_det_27650999452153
// MI455X (gfx1250) — matrix operand order repaired and checked by reading, not compiled, not run on hardware
//
#include <hip/hip_runtime.h>
#include <hip/hip_bf16.h>
#include <math.h>

typedef __attribute__((ext_vector_type(16))) _Float16 v16h;
typedef __attribute__((ext_vector_type(8)))  float    v8f;

#define DEV __device__ __forceinline__

// ---------------------------------------------------------------------------
// WMMA helpers (CDNA5, wave32, 16x16x32 f16 -> f32 accumulate)
// ---------------------------------------------------------------------------
DEV v8f wmma16(v16h a, v16h b, v8f c) {
  // D = A(16x32 f16) * B(32x16 f16) + C(16x16 f32)
  return __builtin_amdgcn_wmma_f32_16x16x32_f16(
      /*neg_a=*/false, a, /*neg_b=*/false, b,
      /*c_mod=*/(short)0, c, /*reuse_a=*/false, /*reuse_b=*/false);
}

// A operand: 16(M) x 32(K) f16 built from f32 source p[m*ms + k*ks].
// Layout (ISA 7.12.2): lane<16 -> M=lane, K pairs {0..7,16..23}; lane>=16 -> K {8..15,24..31}.
// Branchless: clamp index to 0 (always valid) and zero by select -> no exec-masked loads.
DEV v16h load_A(const float* p, int ms, int ks, int kmax) {
  int lane = threadIdx.x & 31;
  int m = lane & 15;
  int half = lane >> 4;
  v16h a;
#pragma unroll
  for (int v = 0; v < 8; ++v) {
    int kb = ((v & 4) ? 16 : 0) + half * 8 + (v & 3) * 2;
    bool ok0 = (kb     < kmax);
    bool ok1 = (kb + 1 < kmax);
    float x0 = p[m * ms + (ok0 ? kb     : 0) * ks];
    float x1 = p[m * ms + (ok1 ? kb + 1 : 0) * ks];
    x0 = ok0 ? x0 : 0.0f;
    x1 = ok1 ? x1 : 0.0f;
    a[2 * v]     = (_Float16)x0;
    a[2 * v + 1] = (_Float16)x1;
  }
  return a;
}

// B operand: 32(K) x 16(N) f16 from f32 source p[k*ks + n*ns], zero pad k>=kmax / n>=nmax.
// Layout: lane = N (mod 16); lane<16 -> K {0..7,16..23}; lane>=16 -> K {8..15,24..31} (matches load_A).
DEV v16h load_B(const float* p, int ks, int ns, int kmax, int nmax) {
  int lane = threadIdx.x & 31;
  int n = lane & 15;
  int kh = lane >> 4;
  int nc = (n < nmax) ? n : 0;
  bool okn = (n < nmax);
  v16h b;
#pragma unroll
  for (int v = 0; v < 8; ++v) {
    int k = ((v & 4) ? 16 : 0) + kh * 8 + (v & 3) * 2;
    bool ok0 = okn && (k     < kmax);
    bool ok1 = okn && (k + 1 < kmax);
    float x0 = p[((k     < kmax) ? k     : 0) * ks + nc * ns];
    float x1 = p[((k + 1 < kmax) ? k + 1 : 0) * ks + nc * ns];
    x0 = ok0 ? x0 : 0.0f;
    x1 = ok1 ? x1 : 0.0f;
    b[2 * v]     = (_Float16)x0;
    b[2 * v + 1] = (_Float16)x1;
  }
  return b;
}

// C/D: lane = N (mod 16); VGPR r holds M = r + 8*(lane>>4). Store 16x16 f32 tile.
DEV void store_CD(float* p, int ms, v8f c) {
  int lane = threadIdx.x & 31;
  int n = lane & 15;
  int half = lane >> 4;
#pragma unroll
  for (int r = 0; r < 8; ++r) p[(r + half * 8) * ms + n] = c[r];
}

DEV v8f bc8(float x) {
  v8f c;
#pragma unroll
  for (int r = 0; r < 8; ++r) c[r] = x;
  return c;
}
DEV v8f relu8(v8f x) {
#pragma unroll
  for (int r = 0; r < 8; ++r) x[r] = fmaxf(x[r], 0.0f);
  return x;
}
DEV v8f elu8(v8f x) {
#pragma unroll
  for (int r = 0; r < 8; ++r) x[r] = (x[r] > 0.0f) ? x[r] : (__expf(x[r]) - 1.0f);
  return x;
}
DEV float sigm(float x) { return 1.0f / (1.0f + __expf(-x)); }

// ---------------------------------------------------------------------------
// Kernel 0: per-object embedders (scalar; tiny: 3 MLPs of width 16 over B=1024)
// obj layout: obj[(slot*B + b)*16 + k], slot 0..6
// ---------------------------------------------------------------------------
__global__ void __launch_bounds__(256)
obj_kernel(const float* __restrict__ envs,
           const float* __restrict__ W0, const float* __restrict__ b0,
           const float* __restrict__ Wh, const float* __restrict__ bh,
           const float* __restrict__ Wo, const float* __restrict__ bo,
           const float* __restrict__ nullE, float* __restrict__ obj, int B) {
  int b = blockIdx.x * blockDim.x + threadIdx.x;
  if (b >= B) return;
  for (int s = 0; s < 3; ++s) {
    float x[3];
    for (int j = 0; j < 3; ++j) x[j] = envs[b * 9 + s * 3 + j];
    float h[16], h2[16];
    for (int n = 0; n < 16; ++n) {
      float a = b0[s * 16 + n];
      for (int j = 0; j < 3; ++j) a += x[j] * W0[s * 48 + j * 16 + n];
      h[n] = (a > 0.f) ? a : (__expf(a) - 1.f);
    }
    for (int L = 0; L < 3; ++L) {
      for (int n = 0; n < 16; ++n) {
        float a = bh[s * 48 + L * 16 + n];
        for (int k = 0; k < 16; ++k) a += h[k] * Wh[s * 768 + L * 256 + k * 16 + n];
        h2[n] = (a > 0.f) ? a : (__expf(a) - 1.f);
      }
      for (int n = 0; n < 16; ++n) h[n] = h2[n];
    }
    for (int n = 0; n < 16; ++n) {
      float a = bo[s * 16 + n];
      for (int k = 0; k < 16; ++k) a += h[k] * Wo[s * 256 + k * 16 + n];
      obj[(s * B + b) * 16 + n] = a;
    }
  }
  for (int s = 0; s < 4; ++s)
    for (int n = 0; n < 16; ++n)
      obj[((3 + s) * B + b) * 16 + n] = nullE[s * 16 + n];
}

// ---------------------------------------------------------------------------
// Kernel 1: state embedder + attention scores + flipped softmax.
// One wave per block; each block handles tilesPerBlk 16-row tiles of (T,B).
// ---------------------------------------------------------------------------
__global__ void __launch_bounds__(32, 1)
pre_kernel(const float* __restrict__ states, const float* __restrict__ obj,
           const float* __restrict__ seW0, const float* __restrict__ seb0,
           const float* __restrict__ seWh, const float* __restrict__ sebh,
           const float* __restrict__ seWo, const float* __restrict__ sebo,
           const float* __restrict__ attW0, const float* __restrict__ attb0,
           const float* __restrict__ attWh, const float* __restrict__ attbh,
           const float* __restrict__ attWo, const float* __restrict__ attbo,
           float* __restrict__ se_ws, float* __restrict__ cs_ws,
           int B, int T, int tilesPerBlk) {
  __shared__ float ldsT[16 * 16];
  __shared__ float ldsSE[16 * 16];
  __shared__ float ldsX[16 * 32];
  int lane = threadIdx.x & 31;
  int n = lane & 15, half = lane >> 4;

  // persistent weight B-operands (f32 -> f16, K padded to 32)
  v16h BseW0 = load_B(seW0, 16, 1, 4, 16);
  v16h BseWh[3];
  for (int i = 0; i < 3; ++i) BseWh[i] = load_B(seWh + i * 256, 16, 1, 16, 16);
  v16h BseWo  = load_B(seWo, 16, 1, 16, 16);
  v16h BattW0 = load_B(attW0, 16, 1, 32, 16);
  v16h BattWh[2];
  for (int i = 0; i < 2; ++i) BattWh[i] = load_B(attWh + i * 256, 16, 1, 16, 16);
  float bse0 = seb0[n];
  float bseh[3] = {sebh[n], sebh[16 + n], sebh[32 + n]};
  float bseo = sebo[n];
  float batt0 = attb0[n];
  float batth[2] = {attbh[n], attbh[16 + n]};
  float battO = attbo[0];
  float attWoR[16];
#pragma unroll
  for (int k = 0; k < 16; ++k) attWoR[k] = attWo[k];

  int TILES = (T * B) >> 4;
  int BT = B >> 4;
  for (int it = 0; it < tilesPerBlk; ++it) {
    int tile = blockIdx.x * tilesPerBlk + it;
    if (tile >= TILES) break;
    int t = tile / BT;
    int bb = (tile % BT) << 4;

    // ---- state embedder: 4 -> 16 -> 16 -> 16 -> 16 -> 16 (relu, linear out)
    v16h a = load_A(states + (bb * T + t) * 4, T * 4, 1, 4);
    v8f c = wmma16(a, BseW0, bc8(bse0));
    c = relu8(c);
    for (int i = 0; i < 3; ++i) {
      store_CD(ldsT, 16, c);
      c = relu8(wmma16(load_A(ldsT, 16, 1, 16), BseWh[i], bc8(bseh[i])));
    }
    store_CD(ldsT, 16, c);
    c = wmma16(load_A(ldsT, 16, 1, 16), BseWo, bc8(bseo)); // linear output
    store_CD(se_ws + ((size_t)t * B + bb) * 16, 16, c);
    store_CD(ldsSE, 16, c);

    // ---- attention scores for 7 slots
    float sc[7];
    for (int s = 0; s < 7; ++s) {
      for (int idx = lane; idx < 16 * 32; idx += 32) {
        int m = idx >> 5, j = idx & 31;
        ldsX[idx] = (j < 16) ? ldsSE[m * 16 + j]
                             : obj[((size_t)s * B + bb + m) * 16 + (j - 16)];
      }
      v8f h = elu8(wmma16(load_A(ldsX, 32, 1, 32), BattW0, bc8(batt0)));
      for (int i = 0; i < 2; ++i) {
        store_CD(ldsT, 16, h);
        h = elu8(wmma16(load_A(ldsT, 16, 1, 16), BattWh[i], bc8(batth[i])));
      }
      store_CD(ldsT, 16, h);
      float acc = battO;
#pragma unroll
      for (int k = 0; k < 16; ++k) acc += ldsT[n * 16 + k] * attWoR[k];
      sc[s] = acc; // lanes>=16 duplicate lane-16's row, harmless
    }
    // flipped softmax over slots, per row n (lanes<16 write)
    float mx = sc[0];
    for (int s = 1; s < 7; ++s) mx = fmaxf(mx, sc[s]);
    float e[7], sum = 0.f;
    for (int s = 0; s < 7; ++s) { e[s] = __expf(sc[6 - s] - mx); sum += e[s]; }
    if (half == 0) {
      float inv = 1.0f / sum;
      for (int s = 0; s < 7; ++s)
        cs_ws[(size_t)s * T * B + (size_t)t * B + bb + n] = e[s] * inv;
    }
  }
}

// ---------------------------------------------------------------------------
// Kernel 2: fused 2-layer GRU scan. One wave per 16-batch tile; the object
// part of layer-0's input projection is time-invariant -> folded into a
// constant accumulator once; 12 WMMAs per time step.
// ---------------------------------------------------------------------------
DEV v8f gru_gate(const v8f* xi, const v8f* hh, v8f h) {
  v8f out;
#pragma unroll
  for (int r = 0; r < 8; ++r) {
    float rr = sigm(xi[0][r] + hh[0][r]);
    float zz = sigm(xi[1][r] + hh[1][r]);
    float nn = tanhf(xi[2][r] + rr * hh[2][r]);
    out[r] = (1.0f - zz) * nn + zz * h[r];
  }
  return out;
}

__global__ void __launch_bounds__(32, 1)
scan_kernel(const float* __restrict__ se_ws, const float* __restrict__ obj,
            const float* __restrict__ Wi0, const float* __restrict__ Wh0,
            const float* __restrict__ bi0, const float* __restrict__ bh0,
            const float* __restrict__ Wi1, const float* __restrict__ Wh1,
            const float* __restrict__ bi1, const float* __restrict__ bh1,
            const float* __restrict__ initH, float* __restrict__ gru_ws,
            int B, int T) {
  __shared__ float ldsT[16 * 16];
  __shared__ float ldsO[16 * 128];
  int lane = threadIdx.x & 31;
  int n = lane & 15;
  int bb = blockIdx.x << 4;

  // compose full gru input's object part: x_in[m][j], j<16 zero (se slot), j>=16 obj
  for (int idx = lane; idx < 16 * 128; idx += 32) {
    int m = idx >> 7, j = idx & 127;
    ldsO[idx] = (j < 16) ? 0.0f
                         : obj[(((size_t)((j - 16) >> 4)) * B + bb + m) * 16 + ((j - 16) & 15)];
  }

  // constant part of xi0 (bias + obj columns), accumulated over 4 K-chunks of 32
  v8f xiobj[3];
  for (int g = 0; g < 3; ++g) xiobj[g] = bc8(bi0[g * 16 + n]);
  for (int kc = 0; kc < 4; ++kc) {
    v16h ao = load_A(ldsO + kc * 32, 128, 1, 32);
    for (int g = 0; g < 3; ++g) {
      v16h bw = load_B(Wi0 + (size_t)g * 16 * 128 + kc * 32, 1, 128, 32, 16); // B[k][n]=Wi0[n][k]
      xiobj[g] = wmma16(ao, bw, xiobj[g]);
    }
  }

  // persistent recurrent weights
  v16h BWi0se[3], BWh0[3], BWi1[3], BWh1[3];
  float bh0b[3], bi1b[3], bh1b[3];
  for (int g = 0; g < 3; ++g) {
    BWi0se[g] = load_B(Wi0 + (size_t)g * 16 * 128, 1, 128, 16, 16); // first 16 K of Wi0.T
    BWh0[g]   = load_B(Wh0 + g * 16 * 16, 1, 16, 16, 16);
    BWi1[g]   = load_B(Wi1 + g * 16 * 16, 1, 16, 16, 16);
    BWh1[g]   = load_B(Wh1 + g * 16 * 16, 1, 16, 16, 16);
    bh0b[g] = bh0[g * 16 + n];
    bi1b[g] = bi1[g * 16 + n];
    bh1b[g] = bh1[g * 16 + n];
  }

  v8f h0 = bc8(initH[n]);
  v8f h1 = bc8(initH[16 + n]);

  for (int t = 0; t < T; ++t) {
    const float* seP = se_ws + ((size_t)t * B + bb) * 16;
    if (t + 1 < T) __builtin_prefetch(se_ws + ((size_t)(t + 1) * B + bb) * 16, 0, 1);
    v16h aSe = load_A(seP, 16, 1, 16);

    // layer 0
    store_CD(ldsT, 16, h0);
    v16h aH = load_A(ldsT, 16, 1, 16);
    v8f xi[3], hh[3];
    for (int g = 0; g < 3; ++g) {
      xi[g] = wmma16(aSe, BWi0se[g], xiobj[g]);
      hh[g] = wmma16(aH, BWh0[g], bc8(bh0b[g]));
    }
    h0 = gru_gate(xi, hh, h0);

    // layer 1 (consumes y0 = h0 of this step)
    store_CD(ldsT, 16, h0);
    v16h aY = load_A(ldsT, 16, 1, 16);
    store_CD(ldsT, 16, h1);
    v16h aH1 = load_A(ldsT, 16, 1, 16);
    for (int g = 0; g < 3; ++g) {
      xi[g] = wmma16(aY, BWi1[g], bc8(bi1b[g]));
      hh[g] = wmma16(aH1, BWh1[g], bc8(bh1b[g]));
    }
    h1 = gru_gate(xi, hh, h1);

    store_CD(gru_ws + ((size_t)t * B + bb) * 16, 16, h1);
  }
}

// ---------------------------------------------------------------------------
// Kernel 3: relation MLPs (N padded 8->16 with zero columns) weighted by
// attention, then final MLP 72->4->4->4; scatter to out (B,T,4).
// ---------------------------------------------------------------------------
__global__ void __launch_bounds__(32, 1)
post_kernel(const float* __restrict__ gru_ws, const float* __restrict__ obj,
            const float* __restrict__ cs_ws,
            const float* __restrict__ rW0, const float* __restrict__ rb0,
            const float* __restrict__ rWh, const float* __restrict__ rbh,
            const float* __restrict__ rWo, const float* __restrict__ rbo,
            const float* __restrict__ fW0, const float* __restrict__ fb0,
            const float* __restrict__ fWh, const float* __restrict__ fbh,
            const float* __restrict__ fWo, const float* __restrict__ fbo,
            float* __restrict__ out, int B, int T, int tilesPerBlk) {
  __shared__ float ldsT[16 * 16];
  __shared__ float ldsX[16 * 32];
  __shared__ float ldsF[16 * 96]; // final input 16x72 (padded to 96)
  int lane = threadIdx.x & 31;
  int n = lane & 15, half = lane >> 4;

  v16h BfW0[3];
  for (int ch = 0; ch < 3; ++ch) {
    int kk = 72 - ch * 32; if (kk > 32) kk = 32;
    BfW0[ch] = load_B(fW0 + ch * 32 * 4, 4, 1, kk, 4);
  }
  v16h BfWh = load_B(fWh, 4, 1, 4, 4);
  v16h BfWo = load_B(fWo, 4, 1, 4, 4);
  float bf0 = (n < 4) ? fb0[n] : 0.f;
  float bfh = (n < 4) ? fbh[n] : 0.f;
  float bfo = (n < 4) ? fbo[n] : 0.f;

  int TILES = (T * B) >> 4;
  int BT = B >> 4;
  for (int it = 0; it < tilesPerBlk; ++it) {
    int tile = blockIdx.x * tilesPerBlk + it;
    if (tile >= TILES) break;
    int t = tile / BT;
    int bb = (tile % BT) << 4;
    size_t rbase = (size_t)t * B + bb;

    for (int idx = lane; idx < 16 * 96; idx += 32) ldsF[idx] = 0.0f;
    for (int idx = lane; idx < 16 * 16; idx += 32) {
      int m = idx >> 4, k = idx & 15;
      ldsF[m * 96 + k] = gru_ws[(rbase + m) * 16 + k];
    }

    for (int s = 0; s < 7; ++s) {
      for (int idx = lane; idx < 16 * 32; idx += 32) {
        int m = idx >> 5, j = idx & 31;
        ldsX[idx] = (j < 16) ? ldsF[m * 96 + j]
                             : obj[((size_t)s * B + bb + m) * 16 + (j - 16)];
      }
      v16h Bw0 = load_B(rW0 + s * 32 * 8, 8, 1, 32, 8);
      v8f h = elu8(wmma16(load_A(ldsX, 32, 1, 32), Bw0, bc8((n < 8) ? rb0[s * 8 + n] : 0.f)));
      store_CD(ldsT, 16, h);
      v16h Bwh = load_B(rWh + s * 64, 8, 1, 8, 8);
      h = elu8(wmma16(load_A(ldsT, 16, 1, 8), Bwh, bc8((n < 8) ? rbh[s * 8 + n] : 0.f)));
      store_CD(ldsT, 16, h);
      v16h Bwo = load_B(rWo + s * 64, 8, 1, 8, 8);
      h = wmma16(load_A(ldsT, 16, 1, 8), Bwo, bc8((n < 8) ? rbo[s * 8 + n] : 0.f)); // linear
      if (n < 8) {
#pragma unroll
        for (int r = 0; r < 8; ++r) {
          int m = r + half * 8;
          float w = cs_ws[(size_t)s * T * B + rbase + m];
          ldsF[m * 96 + 16 + s * 8 + n] = h[r] * w;
        }
      }
    }

    // final MLP: 72 -> 4 (elu) -> 4 (elu) -> 4 (linear)
    v8f c = bc8(bf0);
    for (int ch = 0; ch < 3; ++ch) {
      int kk = 72 - ch * 32; if (kk > 32) kk = 32;
      c = wmma16(load_A(ldsF + ch * 32, 96, 1, kk), BfW0[ch], c);
    }
    c = elu8(c);
    store_CD(ldsT, 16, c);
    c = elu8(wmma16(load_A(ldsT, 16, 1, 4), BfWh, bc8(bfh)));
    store_CD(ldsT, 16, c);
    c = wmma16(load_A(ldsT, 16, 1, 4), BfWo, bc8(bfo));

    if (n < 4) {
#pragma unroll
      for (int r = 0; r < 8; ++r) {
        int m = r + half * 8;
        out[(((size_t)(bb + m)) * T + t) * 4 + n] = c[r];
      }
    }
  }
}

// ---------------------------------------------------------------------------
extern "C" void kernel_launch(void* const* d_in, const int* in_sizes, int n_in,
                              void* d_out, int out_size, void* d_ws, size_t ws_size,
                              hipStream_t stream) {
  (void)n_in; (void)out_size; (void)ws_size;
  const float* envs   = (const float*)d_in[0];
  const float* states = (const float*)d_in[1];
  const float* seW0 = (const float*)d_in[2];  const float* seb0 = (const float*)d_in[3];
  const float* seWh = (const float*)d_in[4];  const float* sebh = (const float*)d_in[5];
  const float* seWo = (const float*)d_in[6];  const float* sebo = (const float*)d_in[7];
  const float* oeW0 = (const float*)d_in[8];  const float* oeb0 = (const float*)d_in[9];
  const float* oeWh = (const float*)d_in[10]; const float* oebh = (const float*)d_in[11];
  const float* oeWo = (const float*)d_in[12]; const float* oebo = (const float*)d_in[13];
  const float* nullE = (const float*)d_in[14];
  const float* initH = (const float*)d_in[15];
  const float* attW0 = (const float*)d_in[16]; const float* attb0 = (const float*)d_in[17];
  const float* attWh = (const float*)d_in[18]; const float* attbh = (const float*)d_in[19];
  const float* attWo = (const float*)d_in[20]; const float* attbo = (const float*)d_in[21];
  const float* g0Wi = (const float*)d_in[22]; const float* g0Wh = (const float*)d_in[23];
  const float* g0bi = (const float*)d_in[24]; const float* g0bh = (const float*)d_in[25];
  const float* g1Wi = (const float*)d_in[26]; const float* g1Wh = (const float*)d_in[27];
  const float* g1bi = (const float*)d_in[28]; const float* g1bh = (const float*)d_in[29];
  const float* rW0 = (const float*)d_in[30]; const float* rb0 = (const float*)d_in[31];
  const float* rWh = (const float*)d_in[32]; const float* rbh = (const float*)d_in[33];
  const float* rWo = (const float*)d_in[34]; const float* rbo = (const float*)d_in[35];
  const float* fW0 = (const float*)d_in[36]; const float* fb0 = (const float*)d_in[37];
  const float* fWh = (const float*)d_in[38]; const float* fbh = (const float*)d_in[39];
  const float* fWo = (const float*)d_in[40]; const float* fbo = (const float*)d_in[41];

  const int B = in_sizes[0] / 9;
  const int T = in_sizes[1] / (B * 4);

  float* ws = (float*)d_ws;
  float* obj    = ws;                              // 7*B*16
  float* se_ws  = obj    + (size_t)7 * B * 16;     // T*B*16
  float* cs_ws  = se_ws  + (size_t)T * B * 16;     // 7*T*B
  float* gru_ws = cs_ws  + (size_t)7 * T * B;      // T*B*16
  float* outF   = (float*)d_out;                   // B*T*4

  obj_kernel<<<(B + 255) / 256, 256, 0, stream>>>(envs, oeW0, oeb0, oeWh, oebh, oeWo, oebo,
                                                  nullE, obj, B);

  const int TILES = (T * B) / 16;
  const int TPB = 16;
  pre_kernel<<<TILES / TPB, 32, 0, stream>>>(states, obj, seW0, seb0, seWh, sebh, seWo, sebo,
                                             attW0, attb0, attWh, attbh, attWo, attbo,
                                             se_ws, cs_ws, B, T, TPB);

  scan_kernel<<<B / 16, 32, 0, stream>>>(se_ws, obj, g0Wi, g0Wh, g0bi, g0bh,
                                         g1Wi, g1Wh, g1bi, g1bh, initH, gru_ws, B, T);

  post_kernel<<<TILES / TPB, 32, 0, stream>>>(gru_ws, obj, cs_ws,
                                              rW0, rb0, rWh, rbh, rWo, rbo,
                                              fW0, fb0, fWh, fbh, fWo, fbo,
                                              outF, B, T, TPB);
}
